// RingNN_59880434041547
// MI455X (gfx1250) — matrix operand order repaired and checked by reading, not compiled, not run on hardware
//
#include <hip/hip_runtime.h>
#include <hip/hip_bf16.h>

typedef __attribute__((ext_vector_type(16))) _Float16 v16h;
typedef __attribute__((ext_vector_type(8)))  _Float16 v8h;
typedef __attribute__((ext_vector_type(8)))  float    v8f;

// ---------------- workspace layout ----------------
// F   : [4096][576] _Float16   (cos||sin features of h2)   4,718,592 B
// WT  : [32][576]   _Float16   (FF weight matrix, transposed, zero-padded)
// w1sc: [16]  float2 (cos,sin of conv1 weights)
// w2sc: [512] float2 (cos,sin of conv2 weights)
#define F_OFF    0
#define WT_OFF   4718592
#define W1_OFF   (WT_OFF + 36864)
#define W2_OFF   (W1_OFF + 128)

// ---------------- kernel 1: weight prep ----------------
__global__ void ring_prep_kernel(const float* __restrict__ conv1_w,
                                 const float* __restrict__ conv2_w,
                                 const float* __restrict__ ff_w,
                                 float2* __restrict__ w1sc,
                                 float2* __restrict__ w2sc,
                                 _Float16* __restrict__ WT) {
    int tid = threadIdx.x;              // 512 threads, 1 block
    if (tid < 16)  { float s, c; sincosf(conv1_w[tid], &s, &c); w1sc[tid] = make_float2(c, s); }
    if (tid < 512) { float s, c; sincosf(conv2_w[tid], &s, &c); w2sc[tid] = make_float2(c, s); }
    // WT[row][k], row<10: dir_x col j=row -> [cos w | sin w]
    //             row in 10..19: dir_y col j=row-10 -> [-sin w | cos w]
    for (int t = tid; t < 32 * 576; t += 512) {
        int row = t / 576, k = t % 576;
        float val = 0.0f;
        if (row < 20) {
            int j  = row % 10;
            int kk = k % 288;
            float s, c;
            sincosf(ff_w[kk * 10 + j], &s, &c);
            bool hi = (k >= 288);               // sin-feature half of A
            val = (row < 10) ? (hi ? s : c)     // dir_x
                             : (hi ? c : -s);   // dir_y
        }
        WT[t] = (_Float16)val;
    }
}

// ---------------- kernel 2: fused ring conv1 + conv2 (one block per image) ----------------
__global__ void __launch_bounds__(128)
ring_conv_kernel(const float* __restrict__ x,
                 const float2* __restrict__ w1sc,
                 const float2* __restrict__ w2sc,
                 _Float16* __restrict__ F) {
    __shared__ float  c0[784], s0[784];     // input sincos
    __shared__ float  c1[784], s1[784];     // h1 (cos,sin), layout [oc(4)][14][14]
    __shared__ float2 w2s[512];
    __shared__ float2 w1s[16];

    const int img = blockIdx.x;
    const int tid = threadIdx.x;
    const float* xi = x + img * 784;

    for (int t = tid; t < 784; t += 128) {
        float s, c;
        sincosf(xi[t], &s, &c);
        c0[t] = c; s0[t] = s;
    }
    for (int t = tid; t < 512; t += 128) w2s[t] = w2sc[t];
    if (tid < 16) w1s[tid] = w1sc[tid];
    __syncthreads();

    // conv1: 2x2 stride 2, 4 out channels; 784 outputs
    for (int t = tid; t < 784; t += 128) {
        int oc = t / 196, pos = t % 196;
        int i = pos / 14, j = pos % 14;
        float X = 0.f, Y = 0.f;
        #pragma unroll
        for (int dy = 0; dy < 2; ++dy)
            #pragma unroll
            for (int dx = 0; dx < 2; ++dx) {
                int pi = (2 * i + dy) * 28 + (2 * j + dx);
                float c = c0[pi], s = s0[pi];
                float2 w = w1s[oc * 4 + dy * 2 + dx];   // (cos w, sin w)
                X += c * w.x + s * w.y;                 // cos(p-w)
                Y += s * w.x - c * w.y;                 // sin(p-w)
            }
        float r = rsqrtf(X * X + Y * Y + 1e-30f);
        c1[t] = X * r; s1[t] = Y * r;                   // cos/sin of atan2(Y,X)
    }
    __syncthreads();

    // conv2: 4x4 stride 2 over [4][14][14] -> [6][6], 8 out channels; 288 outputs
    _Float16* Fi = F + img * 576;
    for (int t = tid; t < 288; t += 128) {
        int oc = t / 36, pos = t % 36;
        int i = pos / 6, j = pos % 6;
        float X = 0.f, Y = 0.f;
        for (int cin = 0; cin < 4; ++cin)
            #pragma unroll
            for (int kh = 0; kh < 4; ++kh)
                #pragma unroll
                for (int kw = 0; kw < 4; ++kw) {
                    int hi = cin * 196 + (2 * i + kh) * 14 + (2 * j + kw);
                    float c = c1[hi], s = s1[hi];
                    float2 w = w2s[((oc * 4 + cin) * 4 + kh) * 4 + kw];
                    X += c * w.x + s * w.y;
                    Y += s * w.x - c * w.y;
                }
        float r = rsqrtf(X * X + Y * Y + 1e-30f);
        int fidx = (i * 6 + j) * 8 + oc;                // matches h2.reshape(B,288) order
        Fi[fidx]       = (_Float16)(X * r);             // cos half (k < 288)
        Fi[288 + fidx] = (_Float16)(Y * r);             // sin half
    }
}

// ---------------- kernel 3: FF layer as WMMA GEMM ----------------
// OUT[4096][10]: dir = F(4096x576) * WT^T(576x32); out = dy * rsqrt(dx^2+dy^2)
__global__ void __launch_bounds__(128)
ring_ff_wmma_kernel(const _Float16* __restrict__ F,
                    const _Float16* __restrict__ WT,
                    float* __restrict__ out) {
    __shared__ float Dsh[4][16][26];        // [wave][row][col<20], padded

    const int wave  = threadIdx.x >> 5;
    const int lane  = threadIdx.x & 31;
    const int l15   = lane & 15;
    const int lhalf = lane >> 4;            // 0 or 1
    const int row0  = (blockIdx.x * 4 + wave) * 16;

    // A fragment (16x32 f16, contiguous k order, same as B):
    //   lane half 0: K = kb + 0..15 ; half 1: K = kb+16..31
    const _Float16* Arow  = F + (size_t)(row0 + l15) * 576 + lhalf * 16;
    // B fragment (32x16 f16): lane n=l15, K = kb + 16*lhalf + 0..15 ; B[k][n] = WT[n][k]
    const _Float16* B0row = WT + (size_t)l15 * 576 + lhalf * 16;          // N = 0..15
    const _Float16* B1row = WT + (size_t)(16 + l15) * 576 + lhalf * 16;   // N = 16..31

    v8f acc0 = {};
    v8f acc1 = {};

    for (int kb = 0; kb < 576; kb += 32) {
        v8h alo = *(const v8h*)(Arow + kb);
        v8h ahi = *(const v8h*)(Arow + kb + 8);
        v16h a;
        #pragma unroll
        for (int e = 0; e < 8; ++e) { a[e] = alo[e]; a[8 + e] = ahi[e]; }
        v16h b0 = *(const v16h*)(B0row + kb);
        v16h b1 = *(const v16h*)(B1row + kb);
        acc0 = __builtin_amdgcn_wmma_f32_16x16x32_f16(false, a, false, b0,
                                                      (short)0, acc0, false, false);
        acc1 = __builtin_amdgcn_wmma_f32_16x16x32_f16(false, a, false, b1,
                                                      (short)0, acc1, false, false);
    }

    // C/D layout: lanes 0-15: N=lane, M=v ; lanes 16-31: N=lane-16, M=8+v
    #pragma unroll
    for (int v = 0; v < 8; ++v) {
        int m = v + 8 * lhalf;
        Dsh[wave][m][l15] = acc0[v];
        if (l15 < 4) Dsh[wave][m][16 + l15] = acc1[v];   // only N=16..19 nonzero
    }
    __syncthreads();

    // 160 outputs per wave: out = sin(atan2(dy,dx)) = dy * rsqrt(dx^2+dy^2)
    for (int t = lane; t < 160; t += 32) {
        int r = t / 10, j = t % 10;
        float dx = Dsh[wave][r][j];
        float dy = Dsh[wave][r][10 + j];
        out[(size_t)(row0 + r) * 10 + j] = dy * rsqrtf(dx * dx + dy * dy + 1e-30f);
    }
}

// ---------------- launch ----------------
extern "C" void kernel_launch(void* const* d_in, const int* in_sizes, int n_in,
                              void* d_out, int out_size, void* d_ws, size_t ws_size,
                              hipStream_t stream) {
    (void)in_sizes; (void)n_in; (void)out_size; (void)ws_size;
    const float* x       = (const float*)d_in[0];   // [4096,1,28,28]
    const float* conv1_w = (const float*)d_in[1];   // [4,1,2,2]
    const float* conv2_w = (const float*)d_in[2];   // [8,4,4,4]
    const float* ff_w    = (const float*)d_in[3];   // [288,10]
    float* out = (float*)d_out;                     // [4096,10]

    char* ws = (char*)d_ws;
    _Float16* F    = (_Float16*)(ws + F_OFF);
    _Float16* WT   = (_Float16*)(ws + WT_OFF);
    float2*   w1sc = (float2*)(ws + W1_OFF);
    float2*   w2sc = (float2*)(ws + W2_OFF);

    ring_prep_kernel<<<1, 512, 0, stream>>>(conv1_w, conv2_w, ff_w, w1sc, w2sc, WT);
    ring_conv_kernel<<<4096, 128, 0, stream>>>(x, w1sc, w2sc, F);
    ring_ff_wmma_kernel<<<64, 128, 0, stream>>>(F, WT, out);
}
